// HDCAM_24970939859675
// MI455X (gfx1250) — compile-verified, matrix operand order checked by reading, not run on hardware
//
#include <hip/hip_runtime.h>
#include <hip/hip_bf16.h>

// ---------------------------------------------------------------------------
// MI455X (gfx1250) implementation.
// Dominant work: 3x Hopfield (8192x8192x4096) GEMMs -> bf16 WMMA
// (v_wmma_f32_16x16x32_bf16), fp32 accumulate. dm matrix reduces to a
// cumulative mean (diffs clipped to 0 in the tril), so "local" is a scan.
// GEMM takes B pre-transposed (N x K row-major) so that BOTH fragment types
// are K-contiguous in LDS -> all fragment reads are ds_load_b128.
// ---------------------------------------------------------------------------

typedef __bf16 bf16;
typedef __attribute__((ext_vector_type(16))) __bf16 v16bf;
typedef __attribute__((ext_vector_type(8)))  float  v8f;

struct Pack32 { uint4 a, b; };   // 32B container for one v16bf fragment

static __device__ __forceinline__ bf16 f2bf(float f) {
  unsigned u = __builtin_bit_cast(unsigned, f);
  unsigned r = u + 0x7FFFu + ((u >> 16) & 1u);
  unsigned short h = (unsigned short)(r >> 16);
  return __builtin_bit_cast(bf16, h);
}

// ---------------------------------------------------------------------------
// bf16 GEMM: C[M,N] = act(alpha * A@B^T' + bias + beta*Cin, mask)
//   A  (M,K) row-major bf16
//   BT (N,K) row-major bf16  (i.e. logical B (K,N) pre-transposed)
// 128x128 tile per block, 8 wave32 waves; per wave 32x64 = 2x4 wmma tiles,
// K-step 32. A-tile and BT-tile are both 128x32, staged identically in LDS.
// ---------------------------------------------------------------------------
#define TM 128
#define TN 128
#define TK 32
#define LDSK (TK + 8)   // 40 halfs = 80B row stride; 16B-aligned offsets

__global__ __launch_bounds__(256) void k_gemm(
    const bf16* __restrict__ A, long long sA,
    const bf16* __restrict__ BT, long long sB,
    float* __restrict__ C, long long sC,
    bf16* __restrict__ Cbf, long long sCbf,
    const float* __restrict__ Cin, long long sCin,
    const float* __restrict__ bias,
    int M, int N, int Kd, float alpha, float beta, int act, int mask)
{
  (void)M;
  int bz = blockIdx.z;
  A  += (long long)bz * sA;
  BT += (long long)bz * sB;
  if (C)   C   += (long long)bz * sC;
  if (Cbf) Cbf += (long long)bz * sCbf;
  if (Cin) Cin += (long long)bz * sCin;

  __shared__ __align__(16) bf16 As[TM][LDSK];
  __shared__ __align__(16) bf16 Bs[TN][LDSK];

  int tid  = threadIdx.x;
  int wid  = tid >> 5;
  int lane = tid & 31;
  int wrow = (wid & 3) * 32;     // wave row origin within tile
  int wcol = (wid >> 2) * 64;    // wave col origin within tile
  int row0 = blockIdx.y * TM;
  int col0 = blockIdx.x * TN;

  // ---- staging pointers: vector (tid) and (tid+256); both tiles 128x32 ----
  int sr = tid >> 2;             // 0..63
  int sc = (tid & 3) * 8;        // 0,8,16,24
  const bf16* pa0 = A  + (long long)(row0 + sr)      * Kd + sc;
  const bf16* pa1 = A  + (long long)(row0 + 64 + sr) * Kd + sc;
  const bf16* pb0 = BT + (long long)(col0 + sr)      * Kd + sc;
  const bf16* pb1 = BT + (long long)(col0 + 64 + sr) * Kd + sc;
  uint4* da0 = (uint4*)(&As[sr][sc]);
  uint4* da1 = (uint4*)(&As[64 + sr][sc]);
  uint4* db0 = (uint4*)(&Bs[sr][sc]);
  uint4* db1 = (uint4*)(&Bs[64 + sr][sc]);

  // ---- fragment pointers (ISA 7.12.2: per-lane data is K-contiguous) ----
  int m16   = lane & 15;
  int khalf = (lane < 16) ? 0 : 8;
  const uint4 *a_lo[2], *a_hi[2], *b_lo[4], *b_hi[4];
#pragma unroll
  for (int i = 0; i < 2; ++i) {
    const bf16* base = &As[wrow + i * 16 + m16][0];
    a_lo[i] = (const uint4*)(base + khalf);
    a_hi[i] = (const uint4*)(base + khalf + 16);
  }
#pragma unroll
  for (int j = 0; j < 4; ++j) {
    const bf16* base = &Bs[wcol + j * 16 + m16][0];
    b_lo[j] = (const uint4*)(base + khalf);
    b_hi[j] = (const uint4*)(base + khalf + 16);
  }

  v8f acc[2][4];
#pragma unroll
  for (int i = 0; i < 2; ++i)
#pragma unroll
    for (int j = 0; j < 4; ++j)
#pragma unroll
      for (int r = 0; r < 8; ++r) acc[i][j][r] = 0.0f;

  for (int k0 = 0; k0 < Kd; k0 += TK) {
    // gather 4 x 16B from global into regs, then 4 x ds_store_b128
    uint4 ra0 = *(const uint4*)pa0;
    uint4 ra1 = *(const uint4*)pa1;
    uint4 rb0 = *(const uint4*)pb0;
    uint4 rb1 = *(const uint4*)pb1;
    if (k0 + TK < Kd) {   // prefetch next K-tile (global_prefetch_b8)
      __builtin_prefetch(pa0 + TK, 0, 0);
      __builtin_prefetch(pb0 + TK, 0, 0);
    }
    *da0 = ra0; *da1 = ra1; *db0 = rb0; *db1 = rb1;
    pa0 += TK; pa1 += TK; pb0 += TK; pb1 += TK;
    __syncthreads();

    v16bf af[2], bfr[4];
#pragma unroll
    for (int i = 0; i < 2; ++i) {
      Pack32 p; p.a = *a_lo[i]; p.b = *a_hi[i];
      af[i] = __builtin_bit_cast(v16bf, p);
    }
#pragma unroll
    for (int j = 0; j < 4; ++j) {
      Pack32 p; p.a = *b_lo[j]; p.b = *b_hi[j];
      bfr[j] = __builtin_bit_cast(v16bf, p);
    }
#pragma unroll
    for (int i = 0; i < 2; ++i)
#pragma unroll
      for (int j = 0; j < 4; ++j)
        acc[i][j] = __builtin_amdgcn_wmma_f32_16x16x32_bf16(
            false, af[i], false, bfr[j], (short)0, acc[i][j], false, false);
    __syncthreads();
  }

  // ---- epilogue ----
  int cl    = lane & 15;
  int rhalf = (lane < 16) ? 0 : 8;
#pragma unroll
  for (int i = 0; i < 2; ++i)
#pragma unroll
    for (int j = 0; j < 4; ++j) {
      int col = col0 + wcol + j * 16 + cl;
#pragma unroll
      for (int r = 0; r < 8; ++r) {
        int row = row0 + wrow + i * 16 + rhalf + r;
        float v = alpha * acc[i][j][r];
        if (bias) v += bias[col];
        if (Cin)  v += beta * Cin[(long long)row * N + col];
        if (mask == 1) { if (!(col <= row - 4)) v = -65000.0f; }
        if (act == 1)  v = 0.5f * v * (1.0f + erff(v * 0.70710678118f));
        long long off = (long long)row * N + col;
        if (C)   C[off]   = v;
        if (Cbf) Cbf[off] = f2bf(v);
      }
    }
}

// ---------------------------------------------------------------------------
// Row softmax (optionally scaled by exp(*log_scale_ptr)), f32 in -> bf16 out.
// ---------------------------------------------------------------------------
__global__ __launch_bounds__(256) void k_softmax_bf(
    const float* __restrict__ X, bf16* __restrict__ Y, int cols,
    const float* __restrict__ log_scale_ptr)
{
  long long row = blockIdx.x;
  const float* x = X + row * (long long)cols;
  bf16* y = Y + row * (long long)cols;
  float scale = log_scale_ptr ? __expf(*log_scale_ptr) : 1.0f;
  __shared__ float sm[256];
  int tid = threadIdx.x;
  float m = -3.0e38f;
  for (int i = tid; i < cols; i += 256) m = fmaxf(m, scale * x[i]);
  sm[tid] = m; __syncthreads();
  for (int s = 128; s > 0; s >>= 1) { if (tid < s) sm[tid] = fmaxf(sm[tid], sm[tid + s]); __syncthreads(); }
  m = sm[0]; __syncthreads();
  float sum = 0.f;
  for (int i = tid; i < cols; i += 256) sum += __expf(scale * x[i] - m);
  sm[tid] = sum; __syncthreads();
  for (int s = 128; s > 0; s >>= 1) { if (tid < s) sm[tid] += sm[tid + s]; __syncthreads(); }
  float inv = 1.0f / sm[0];
  for (int i = tid; i < cols; i += 256) y[i] = f2bf(__expf(scale * x[i] - m) * inv);
}

// ---------------------------------------------------------------------------
// Trigram HDC binding: hdc = 0.7*chars*roll(c1,1,d)*roll(c2,2,d) + 0.3*chars
// ---------------------------------------------------------------------------
__global__ void k_trigram(const int* __restrict__ idx, const float* __restrict__ cb,
                          float* __restrict__ hdc, int Bv, int Tv, int Dv)
{
  long long i = blockIdx.x * 256LL + threadIdx.x;
  if (i >= (long long)Bv * Tv * Dv) return;
  int d = (int)(i % Dv);
  long long bt = i / Dv;
  int t = (int)(bt % Tv), b = (int)(bt / Tv);
  int i0 = idx[b * Tv + t];
  float ch = cb[(long long)i0 * Dv + d];
  float v = 0.3f * ch;
  if (t >= 2) {
    int i1 = idx[b * Tv + t - 1];
    int i2 = idx[b * Tv + t - 2];
    float a  = cb[(long long)i1 * Dv + ((d + Dv - 1) % Dv)];
    float c2 = cb[(long long)i2 * Dv + ((d + Dv - 2) % Dv)];
    v += 0.7f * ch * a * c2;
  }
  hdc[i] = v;
}

__global__ __launch_bounds__(256) void k_gate(
    const float* __restrict__ hdc, const float* __restrict__ gw,
    const float* __restrict__ gb, float* __restrict__ imp, int Dv)
{
  long long row = blockIdx.x;
  const float* x = hdc + row * (long long)Dv;
  __shared__ float sm[256];
  int tid = threadIdx.x;
  float s = 0.f;
  for (int i = tid; i < Dv; i += 256) s += x[i] * gw[i];
  sm[tid] = s; __syncthreads();
  for (int r = 128; r > 0; r >>= 1) { if (tid < r) sm[tid] += sm[tid + r]; __syncthreads(); }
  if (tid == 0) imp[row] = 1.0f / (1.0f + __expf(-(sm[0] + gb[0])));
}

__global__ void k_gated(const float* __restrict__ hdc, const float* __restrict__ imp,
                        bf16* __restrict__ out, long long n, int Dv)
{
  long long i = blockIdx.x * 256LL + threadIdx.x;
  if (i < n) out[i] = f2bf(hdc[i] * imp[i / Dv]);
}

// ctx0 = 0.88 * cumulative_mean_t(gated) + hdc   (dm == tril(ones)/rowsum)
__global__ void k_ctx0(const bf16* __restrict__ gated, const float* __restrict__ hdc,
                       float* __restrict__ ctx0, int Bv, int Tv, int Dv)
{
  long long i = blockIdx.x * 256LL + threadIdx.x;
  if (i >= (long long)Bv * Dv) return;
  int d = (int)(i % Dv), b = (int)(i / Dv);
  long long base = (long long)b * Tv * Dv + d;
  float acc = 0.f;
  for (int t = 0; t < Tv; ++t) {
    long long o = base + (long long)t * Dv;
    acc += (float)gated[o];
    ctx0[o] = 0.88f * acc / (float)(t + 1) + hdc[o];
  }
}

__global__ void k_convert(const float* __restrict__ x, bf16* __restrict__ y, long long n)
{
  long long i = blockIdx.x * 256LL + threadIdx.x;
  if (i < n) y[i] = f2bf(x[i]);
}

// f32 (R,C) row-major -> bf16 (C,R), batched via grid.z
__global__ __launch_bounds__(256) void k_transpose_bf(
    const float* __restrict__ X, bf16* __restrict__ Y, int R, int Cc,
    long long sX, long long sY)
{
  X += (long long)blockIdx.z * sX;
  Y += (long long)blockIdx.z * sY;
  __shared__ float tile[32][33];
  int c0 = blockIdx.x * 32, r0 = blockIdx.y * 32;
  int tx = threadIdx.x & 31, ty = threadIdx.x >> 5;   // 32x8
  for (int i = 0; i < 32; i += 8) {
    int r = r0 + ty + i, c = c0 + tx;
    tile[ty + i][tx] = (r < R && c < Cc) ? X[(long long)r * Cc + c] : 0.f;
  }
  __syncthreads();
  for (int i = 0; i < 32; i += 8) {
    int c = c0 + ty + i, r = r0 + tx;
    if (c < Cc && r < R) Y[(long long)c * R + r] = f2bf(tile[tx][ty + i]);
  }
}

// bf16 (R,C) row-major -> bf16 (C,R), batched via grid.z
__global__ __launch_bounds__(256) void k_transpose_bb(
    const bf16* __restrict__ X, bf16* __restrict__ Y, int R, int Cc,
    long long sX, long long sY)
{
  X += (long long)blockIdx.z * sX;
  Y += (long long)blockIdx.z * sY;
  __shared__ bf16 tile[32][33];
  int c0 = blockIdx.x * 32, r0 = blockIdx.y * 32;
  int tx = threadIdx.x & 31, ty = threadIdx.x >> 5;   // 32x8
  for (int i = 0; i < 32; i += 8) {
    int r = r0 + ty + i, c = c0 + tx;
    tile[ty + i][tx] = (r < R && c < Cc) ? X[(long long)r * Cc + c] : f2bf(0.f);
  }
  __syncthreads();
  for (int i = 0; i < 32; i += 8) {
    int c = c0 + ty + i, r = r0 + tx;
    if (c < Cc && r < R) Y[(long long)c * R + r] = tile[tx][ty + i];
  }
}

// chunk-8 mean pool of contexts -> bf16 (B,nC,D)
__global__ void k_pool(const float* __restrict__ ctx, bf16* __restrict__ pooled,
                       int Tv, int Dv, int nC, int CHv, long long total)
{
  long long i = blockIdx.x * 256LL + threadIdx.x;
  if (i >= total) return;
  int d = (int)(i % Dv);
  long long bc = i / Dv;
  int c = (int)(bc % nC), b = (int)(bc / nC);
  const float* p = ctx + ((long long)b * Tv + (long long)c * CHv) * Dv + d;
  float s = 0.f;
  for (int j = 0; j < CHv; ++j) s += p[(long long)j * Dv];
  pooled[i] = f2bf(s / (float)CHv);
}

// phrase_ctx = (tril(0.9^(c-s)) row-normalized) @ phrase  -> bf16
__global__ void k_phrasectx(const float* __restrict__ phrase, bf16* __restrict__ out,
                            int nC, int Dv, long long total)
{
  long long i = blockIdx.x * 256LL + threadIdx.x;
  if (i >= total) return;
  int d = (int)(i % Dv);
  long long bc = i / Dv;
  int c = (int)(bc % nC), b = (int)(bc / nC);
  const float* p = phrase + (long long)b * nC * Dv + d;
  float w = 1.0f, acc = 0.f, norm = 0.f;
  for (int s = c; s >= 0; --s) { acc += w * p[(long long)s * Dv]; norm += w; w *= 0.9f; }
  out[i] = f2bf(acc / norm);
}

// concat [f_char | f_phr | res_embed[idx]+pos_embed] + LayerNorm -> bf16 (N,3H)
__global__ __launch_bounds__(256) void k_concat_ln(
    const float* __restrict__ fchar, const float* __restrict__ fphr,
    const float* __restrict__ res_embed, const float* __restrict__ pos_embed,
    const int* __restrict__ idx, const float* __restrict__ lng,
    const float* __restrict__ lnb, bf16* __restrict__ xout,
    int Tv, int Hv, int CHv)
{
  int n = blockIdx.x;
  int t = n % Tv, b = n / Tv;
  int C3 = 3 * Hv;
  __shared__ float rowbuf[1536];
  __shared__ float sm[256], sm2[256];
  const float* fc = fchar + (long long)n * Hv;
  const float* fp = fphr + (long long)(b * (Tv / CHv) + t / CHv) * Hv;
  int id = idx[n];
  int tid = threadIdx.x;
  for (int i = tid; i < Hv; i += 256) {
    rowbuf[i]          = fc[i];
    rowbuf[Hv + i]     = fp[i];
    rowbuf[2 * Hv + i] = res_embed[(long long)id * Hv + i] + pos_embed[(long long)t * Hv + i];
  }
  __syncthreads();
  float s = 0.f, s2 = 0.f;
  for (int i = tid; i < C3; i += 256) { float v = rowbuf[i]; s += v; s2 += v * v; }
  sm[tid] = s; sm2[tid] = s2; __syncthreads();
  for (int r = 128; r > 0; r >>= 1) {
    if (tid < r) { sm[tid] += sm[tid + r]; sm2[tid] += sm2[tid + r]; }
    __syncthreads();
  }
  float mu  = sm[0] / (float)C3;
  float var = sm2[0] / (float)C3 - mu * mu;
  float inv = rsqrtf(var + 1e-5f);
  bf16* yo = xout + (long long)n * C3;
  for (int i = tid; i < C3; i += 256)
    yo[i] = f2bf((rowbuf[i] - mu) * inv * lng[i] + lnb[i]);
}

// ---------------------------------------------------------------------------
extern "C" void kernel_launch(void* const* d_in, const int* in_sizes, int n_in,
                              void* d_out, int out_size, void* d_ws, size_t ws_size,
                              hipStream_t stream)
{
  (void)in_sizes; (void)n_in; (void)out_size; (void)ws_size;
  constexpr int Bc = 8, Tc = 1024, Vc = 256, Dc = 4096, Kc = 8192, Hc = 512, CHc = 8, BNc = 256;
  constexpr long long Nr = (long long)Bc * Tc;   // 8192 rows
  constexpr int nC = Tc / CHc;                   // 128 chunks

  const int*   idx        = (const int*)  d_in[0];
  const float* codebook   = (const float*)d_in[1];
  /* d_in[2] log_decay: dm reduces to cumulative mean -> unused */
  const float* gate_w     = (const float*)d_in[3];
  const float* gate_b     = (const float*)d_in[4];
  const float* q_w        = (const float*)d_in[5];
  const float* q_b        = (const float*)d_in[6];
  const float* k_w        = (const float*)d_in[7];
  const float* k_b        = (const float*)d_in[8];
  const float* phrase_w   = (const float*)d_in[9];
  const float* phrase_b   = (const float*)d_in[10];
  const float* mem_keys   = (const float*)d_in[11];
  const float* mem_values = (const float*)d_in[12];
  const float* log_beta   = (const float*)d_in[13];
  const float* proj1_w    = (const float*)d_in[14];
  const float* proj1_b    = (const float*)d_in[15];
  const float* proj2_w    = (const float*)d_in[16];
  const float* proj2_b    = (const float*)d_in[17];
  const float* phr_proj_w = (const float*)d_in[18];
  const float* phr_proj_b = (const float*)d_in[19];
  const float* res_embed  = (const float*)d_in[20];
  const float* pos_embed  = (const float*)d_in[21];
  const float* ln_g       = (const float*)d_in[22];
  const float* ln_b       = (const float*)d_in[23];
  const float* nav_w0     = (const float*)d_in[24];
  const float* nav_b0     = (const float*)d_in[25];
  const float* nav_w1     = (const float*)d_in[26];
  const float* nav_b1     = (const float*)d_in[27];
  const float* nav_w2     = (const float*)d_in[28];
  const float* nav_b2     = (const float*)d_in[29];
  const float* head_w     = (const float*)d_in[30];
  const float* head_b     = (const float*)d_in[31];
  float* out = (float*)d_out;

  char* ws = (char*)d_ws;
  size_t off = 0;
  auto alloc = [&](size_t bytes) -> void* {
    off = (off + 255) & ~(size_t)255;
    void* p = ws + off;
    off += bytes;
    return p;
  };

  // hdc + ctx0 are contiguous (each 128MB, 256-aligned sizes) and both dead
  // before the Hopfield stage; the 256MB logits buffer S aliases them.
  float* hdc    = (float*)alloc(Nr * Dc * 4);
  float* ctx0   = (float*)alloc(Nr * Dc * 4);
  float* S      = hdc;                                   // (Nr, Kc) f32 alias
  float* imp    = (float*)alloc(Nr * 4);
  bf16*  gated  = (bf16*) alloc(Nr * Dc * 2);
  bf16*  gatedT = (bf16*) alloc(Nr * Dc * 2);            // (B, D, T)
  bf16*  qbf    = (bf16*) alloc(Nr * BNc * 2);
  bf16*  kbf    = (bf16*) alloc(Nr * BNc * 2);
  float* scores = (float*)alloc((long long)Bc * Tc * Tc * 4);
  bf16*  resbf  = (bf16*) alloc((long long)Bc * Tc * Tc * 2);
  float* ctxf   = (float*)alloc(Nr * Dc * 4);
  bf16*  ctxbf  = (bf16*) alloc(Nr * Dc * 2);
  bf16*  mkbf   = (bf16*) alloc((long long)Kc * Dc * 2); // mem_keys (Kc,D)
  bf16*  mkT    = (bf16*) alloc((long long)Dc * Kc * 2); // mem_keys^T (D,Kc)
  bf16*  mvT    = (bf16*) alloc((long long)Hc * Kc * 2); // mem_values^T (H,Kc)
  bf16*  attnbf = (bf16*) alloc(Nr * Kc * 2);
  bf16*  qh0    = (bf16*) alloc(Nr * Dc * 2);
  bf16*  qh1    = (bf16*) alloc(Nr * Dc * 2);
  bf16*  pooled = (bf16*) alloc((long long)Bc * nC * Dc * 2);
  bf16*  phwT   = (bf16*) alloc((long long)Dc * Dc * 2);
  float* phrase = (float*)alloc((long long)Bc * nC * Dc * 4);
  bf16*  phctx  = (bf16*) alloc((long long)Bc * nC * Dc * 2);
  bf16*  phpT   = (bf16*) alloc((long long)Hc * Dc * 2);
  float* fphr   = (float*)alloc((long long)Bc * nC * Hc * 4);
  bf16*  p1T    = (bf16*) alloc((long long)Hc * Dc * 2);
  bf16*  p2T    = (bf16*) alloc((long long)Hc * Hc * 2);
  bf16*  qwT    = (bf16*) alloc((long long)BNc * Dc * 2);
  bf16*  kwT    = (bf16*) alloc((long long)BNc * Dc * 2);
  bf16*  t1     = (bf16*) alloc(Nr * Hc * 2);
  float* retr   = (float*)alloc(Nr * Hc * 4);
  float* fchar  = (float*)alloc(Nr * Hc * 4);
  bf16*  xbf    = (bf16*) alloc(Nr * 3 * Hc * 2);
  bf16*  n0T    = (bf16*) alloc((long long)Hc * 3 * Hc * 2);
  bf16*  n1T    = (bf16*) alloc((long long)Hc * Hc * 2);
  bf16*  n2T    = (bf16*) alloc((long long)Hc * Hc * 2);
  bf16*  hwT    = (bf16*) alloc((long long)Vc * Hc * 2);
  bf16*  h0     = (bf16*) alloc(Nr * Hc * 2);
  bf16*  h1     = (bf16*) alloc(Nr * Hc * 2);
  bf16*  h2     = (bf16*) alloc(Nr * Hc * 2);

  auto cvt = [&](const float* src, bf16* dst, long long n) {
    k_convert<<<dim3((unsigned)((n + 255) / 256)), dim3(256), 0, stream>>>(src, dst, n);
  };
  // f32 (R,C) -> bf16 (C,R): weights (K,N) -> BT (N,K)
  auto twf = [&](const float* src, bf16* dst, int R, int C) {
    k_transpose_bf<<<dim3((unsigned)(C / 32), (unsigned)(R / 32), 1),
                     dim3(256), 0, stream>>>(src, dst, R, C, 0, 0);
  };
  auto gemm = [&](const bf16* A, long long sA, const bf16* BT, long long sB,
                  float* C, long long sC, bf16* Cbf, long long sCbf,
                  const float* Cin, long long sCin, const float* bias,
                  int M, int N, int K, float alpha, float beta,
                  int act, int mask, int batch) {
    dim3 g((unsigned)(N / 128), (unsigned)(M / 128), (unsigned)batch);
    k_gemm<<<g, dim3(256), 0, stream>>>(A, sA, BT, sB, C, sC, Cbf, sCbf,
                                        Cin, sCin, bias, M, N, K, alpha, beta, act, mask);
  };

  // ---- 1. trigram encoding ----
  {
    long long n = Nr * Dc;
    k_trigram<<<dim3((unsigned)((n + 255) / 256)), dim3(256), 0, stream>>>(
        idx, codebook, hdc, Bc, Tc, Dc);
  }
  // ---- 2. gate + gated (bf16) + ctx0 scan ----
  k_gate<<<dim3((unsigned)Nr), dim3(256), 0, stream>>>(hdc, gate_w, gate_b, imp, Dc);
  {
    long long n = Nr * Dc;
    k_gated<<<dim3((unsigned)((n + 255) / 256)), dim3(256), 0, stream>>>(hdc, imp, gated, n, Dc);
  }
  {
    long long n = (long long)Bc * Dc;
    k_ctx0<<<dim3((unsigned)((n + 255) / 256)), dim3(256), 0, stream>>>(gated, hdc, ctx0, Bc, Tc, Dc);
  }
  // per-batch gated^T (T,D)->(D,T) for the resonance GEMM
  k_transpose_bb<<<dim3(Dc / 32, Tc / 32, Bc), dim3(256), 0, stream>>>(
      gated, gatedT, Tc, Dc, (long long)Tc * Dc, (long long)Dc * Tc);

  // ---- 3. weight conversions (all B operands stored as (N,K)) ----
  twf(q_w, qwT, Dc, BNc);
  twf(k_w, kwT, Dc, BNc);
  twf(phrase_w, phwT, Dc, Dc);
  cvt(mem_keys, mkbf, (long long)Kc * Dc);          // BT for S = qh @ mk^T
  twf(mem_keys, mkT, Kc, Dc);                       // BT for qh = attn @ mk
  twf(mem_values, mvT, Kc, Hc);
  twf(proj1_w, p1T, Dc, Hc);
  twf(proj2_w, p2T, Hc, Hc);
  twf(phr_proj_w, phpT, Dc, Hc);
  twf(nav_w0, n0T, 3 * Hc, Hc);
  twf(nav_w1, n1T, Hc, Hc);
  twf(nav_w2, n2T, Hc, Hc);
  twf(head_w, hwT, Hc, Vc);

  // ---- 4. q/k projections + resonance attention ----
  gemm(gated, 0, qwT, 0, nullptr, 0, qbf, 0, nullptr, 0, q_b,
       (int)Nr, BNc, Dc, 1.f, 0.f, 0, 0, 1);
  gemm(gated, 0, kwT, 0, nullptr, 0, kbf, 0, nullptr, 0, k_b,
       (int)Nr, BNc, Dc, 1.f, 0.f, 0, 0, 1);
  // scores = q @ k^T : BT layout == k itself (T,BN)
  gemm(qbf, (long long)Tc * BNc, kbf, (long long)Tc * BNc,
       scores, (long long)Tc * Tc, nullptr, 0, nullptr, 0, nullptr,
       Tc, Tc, BNc, 0.0625f /* BN^-0.5 */, 0.f, 0, /*mask=*/1, Bc);
  k_softmax_bf<<<dim3((unsigned)Nr), dim3(256), 0, stream>>>(scores, resbf, Tc, nullptr);
  // contexts = 0.12*(resonance @ gated) + (0.88*local + hdc)
  gemm(resbf, (long long)Tc * Tc, gatedT, (long long)Dc * Tc,
       ctxf, (long long)Tc * Dc, ctxbf, (long long)Tc * Dc,
       ctx0, (long long)Tc * Dc, nullptr, Tc, Dc, Tc, 0.12f, 1.f, 0, 0, Bc);

  // ---- 5. Hopfield retrieval (dominant WMMA work) ----
  const bf16* qcur = ctxbf;
  for (int it = 0; it < 3; ++it) {
    gemm(qcur, 0, mkbf, 0, S, 0, nullptr, 0, nullptr, 0, nullptr,
         (int)Nr, Kc, Dc, 1.f, 0.f, 0, 0, 1);
    k_softmax_bf<<<dim3((unsigned)Nr), dim3(256), 0, stream>>>(S, attnbf, Kc, log_beta);
    if (it == 0) {
      gemm(attnbf, 0, mkT, 0, nullptr, 0, qh0, 0, nullptr, 0, nullptr,
           (int)Nr, Dc, Kc, 1.f, 0.f, 0, 0, 1);
      qcur = qh0;
    } else if (it == 1) {
      gemm(attnbf, 0, mkT, 0, nullptr, 0, qh1, 0, nullptr, 0, nullptr,
           (int)Nr, Dc, Kc, 1.f, 0.f, 0, 0, 1);
      qcur = qh1;
    }
  }
  gemm(attnbf, 0, mvT, 0, retr, 0, nullptr, 0, nullptr, 0, nullptr,
       (int)Nr, Hc, Kc, 1.f, 0.f, 0, 0, 1);

  // ---- 6. phrase path ----
  {
    long long n = (long long)Bc * nC * Dc;
    k_pool<<<dim3((unsigned)((n + 255) / 256)), dim3(256), 0, stream>>>(
        ctxf, pooled, Tc, Dc, nC, CHc, n);
  }
  gemm(pooled, 0, phwT, 0, phrase, 0, nullptr, 0, nullptr, 0, phrase_b,
       Bc * nC, Dc, Dc, 1.f, 0.f, 0, 0, 1);
  {
    long long n = (long long)Bc * nC * Dc;
    k_phrasectx<<<dim3((unsigned)((n + 255) / 256)), dim3(256), 0, stream>>>(
        phrase, phctx, nC, Dc, n);
  }
  gemm(phctx, 0, phpT, 0, fphr, 0, nullptr, 0, nullptr, 0, phr_proj_b,
       Bc * nC, Hc, Dc, 1.f, 0.f, 0, 0, 1);

  // ---- 7. f_char = gelu(ctx@W1+b1)@W2+b2 + retrieved ----
  gemm(ctxbf, 0, p1T, 0, nullptr, 0, t1, 0, nullptr, 0, proj1_b,
       (int)Nr, Hc, Dc, 1.f, 0.f, /*gelu*/1, 0, 1);
  gemm(t1, 0, p2T, 0, fchar, 0, nullptr, 0, retr, 0, proj2_b,
       (int)Nr, Hc, Hc, 1.f, 1.f, 0, 0, 1);

  // ---- 8. concat + LayerNorm ----
  k_concat_ln<<<dim3((unsigned)Nr), dim3(256), 0, stream>>>(
      fchar, fphr, res_embed, pos_embed, idx, ln_g, ln_b, xbf, Tc, Hc, CHc);

  // ---- 9. navigator + head ----
  gemm(xbf, 0, n0T, 0, nullptr, 0, h0, 0, nullptr, 0, nav_b0,
       (int)Nr, Hc, 3 * Hc, 1.f, 0.f, 1, 0, 1);
  gemm(h0, 0, n1T, 0, nullptr, 0, h1, 0, nullptr, 0, nav_b1,
       (int)Nr, Hc, Hc, 1.f, 0.f, 1, 0, 1);
  gemm(h1, 0, n2T, 0, nullptr, 0, h2, 0, nullptr, 0, nav_b2,
       (int)Nr, Hc, Hc, 1.f, 0.f, 1, 0, 1);
  gemm(h2, 0, hwT, 0, out, 0, nullptr, 0, nullptr, 0, head_b,
       (int)Nr, Vc, Hc, 1.f, 0.f, 0, 0, 1);
}
